// SelfAttention_81063212744846
// MI455X (gfx1250) — matrix operand order repaired and checked by reading, not compiled, not run on hardware
//
#include <hip/hip_runtime.h>
#include <math.h>

// ---------------------------------------------------------------------------
// Self-attention (SAGAN) for MI455X / gfx1250, wave32, bf16 WMMA 16x16x32.
// Flash-style: the (B,N,N)=256MB attention matrix is never materialized.
// Round 2: fatter waves (32 rows / 32 cols per wave) -> 2x fewer redundant
// L2 reads per WMMA in the two hot kernels.
// ---------------------------------------------------------------------------

typedef __attribute__((ext_vector_type(16))) __bf16 v16bf;
typedef __attribute__((ext_vector_type(8)))  __bf16 v8bf;
typedef __attribute__((ext_vector_type(8)))  float  v8f;

#define BB  4
#define CC  512
#define CP  64
#define NN  4096

__device__ __forceinline__ v8f wmma_bf16(v16bf a, v16bf b, v8f c) {
  return __builtin_amdgcn_wmma_f32_16x16x32_bf16(false, a, false, b, (short)0, c,
                                                 false, false);
}

// A-fragment (16x32 MxK): lane L holds row M=L%16; K subset is two 8-element
// chunks at {h8, 8+h8}, h8 = (L/16)*16.  Source must be row-major with K fast.
__device__ __forceinline__ v16bf load_fragA(const __bf16* __restrict__ base,
                                            int row, int ld, int kbase, int lane) {
  const __bf16* p = base + (size_t)row * ld + kbase;
  int h8 = (lane >> 4) << 4;
  v8bf c0 = *reinterpret_cast<const v8bf*>(p + h8);
  v8bf c1 = *reinterpret_cast<const v8bf*>(p + 8 + h8);
  v16bf r;
#pragma unroll
  for (int e = 0; e < 8; ++e) { r[e] = c0[e]; r[e + 8] = c1[e]; }
  return r;
}

// B-fragment (32x16 KxN): lane L holds column N=L%16; elements e = K half*16+e
// (contiguous 16 values) -> a single 32-byte load from (rows=N, cols=K) bf16.
__device__ __forceinline__ v16bf load_fragB(const __bf16* __restrict__ base,
                                            int row, int ld, int kbase, int lane) {
  const __bf16* p = base + (size_t)row * ld + kbase + ((lane >> 4) << 4);
  return *reinterpret_cast<const v16bf*>(p);
}

// ------------------------- spectral norm sigma ------------------------------
// v = W^T u  =>  u^T W v = ||v||^2, so sigma = ||W^T u||^2 / (||W^T u|| + eps).
__global__ void sn_sigma_kernel(const float* __restrict__ Wf, const float* __restrict__ Wg,
                                const float* __restrict__ Wh, const float* __restrict__ Wv,
                                const float* __restrict__ uf, const float* __restrict__ ug,
                                const float* __restrict__ uh, const float* __restrict__ uv,
                                float* __restrict__ sig_inv) {
  __shared__ float red[256];
  int w = blockIdx.x;
  const float* W; const float* u; int R, Cc;
  if (w == 0)      { W = Wf; u = uf; R = CP; Cc = CC; }
  else if (w == 1) { W = Wg; u = ug; R = CP; Cc = CC; }
  else if (w == 2) { W = Wh; u = uh; R = CP; Cc = CC; }
  else             { W = Wv; u = uv; R = CC; Cc = CP; }
  float ssq = 0.f;
  for (int c = threadIdx.x; c < Cc; c += blockDim.x) {
    float v = 0.f;
    for (int r = 0; r < R; ++r) v += W[r * Cc + c] * u[r];
    ssq += v * v;
  }
  red[threadIdx.x] = ssq;
  __syncthreads();
  for (int s = 128; s > 0; s >>= 1) {
    if ((int)threadIdx.x < s) red[threadIdx.x] += red[threadIdx.x + s];
    __syncthreads();
  }
  if (threadIdx.x == 0) {
    float t = red[0];
    float sigma = t / (sqrtf(t) + 1e-12f);
    sig_inv[w] = 1.0f / sigma;
  }
}

// ------------------------- Wv -> normalized bf16 ----------------------------
__global__ void wv_conv_kernel(const float* __restrict__ Wv,
                               const float* __restrict__ sig_inv,
                               __bf16* __restrict__ Wv_b) {
  int i = blockIdx.x * blockDim.x + threadIdx.x;
  if (i < CC * CP) Wv_b[i] = static_cast<__bf16>(Wv[i] * sig_inv[3]);
}

// ------------------------- fused projections --------------------------------
// Qt[b][n][cp], Kt[b][n][cp]  (row-major, cp fast)  -- A & B frag friendly
// Hm[b][cp][n]                (row-major, n fast)
__global__ void proj_kernel(const float* __restrict__ x,
                            const float* __restrict__ Wf, const float* __restrict__ Wg,
                            const float* __restrict__ Wh,
                            const float* __restrict__ sig_inv,
                            __bf16* __restrict__ Qt, __bf16* __restrict__ Kt,
                            __bf16* __restrict__ Hm) {
  __shared__ float xs[4][64];
  __shared__ float wf_s[4][64], wg_s[4][64], wh_s[4][64];
  int b  = blockIdx.y;
  int j0 = blockIdx.x * 64;
  int tx = threadIdx.x & 63;       // spatial j within tile
  int ty = threadIdx.x >> 6;       // 0..3
  float fa[16], ga[16], ha[16];
#pragma unroll
  for (int p = 0; p < 16; ++p) { fa[p] = 0.f; ga[p] = 0.f; ha[p] = 0.f; }
  for (int c0 = 0; c0 < CC; c0 += 4) {
    __syncthreads();
    xs[ty][tx]   = x[((size_t)b * CC + (c0 + ty)) * NN + j0 + tx];
    wf_s[ty][tx] = Wf[tx * CC + c0 + ty];
    wg_s[ty][tx] = Wg[tx * CC + c0 + ty];
    wh_s[ty][tx] = Wh[tx * CC + c0 + ty];
    __syncthreads();
#pragma unroll
    for (int cc = 0; cc < 4; ++cc) {
      float xv = xs[cc][tx];
#pragma unroll
      for (int p = 0; p < 16; ++p) {
        int cp = ty * 16 + p;
        fa[p] += wf_s[cc][cp] * xv;
        ga[p] += wg_s[cc][cp] * xv;
        ha[p] += wh_s[cc][cp] * xv;
      }
    }
  }
  float sf = sig_inv[0], sg = sig_inv[1], sh = sig_inv[2];
#pragma unroll
  for (int p = 0; p < 16; ++p) {
    int cp = ty * 16 + p;
    Qt[((size_t)b * NN + j0 + tx) * CP + cp] = static_cast<__bf16>(fa[p] * sf);
    Kt[((size_t)b * NN + j0 + tx) * CP + cp] = static_cast<__bf16>(ga[p] * sg);
    Hm[((size_t)b * CP + cp) * NN + j0 + tx] = static_cast<__bf16>(ha[p] * sh);
  }
}

// ------------------------- softmax row stats --------------------------------
// One wave per 32-row block (2 row-tiles held in registers).  Each K-column
// B-fragment pair now feeds 4 WMMAs.  sweep1 = rowmax, sweep2 = sum exp.
__global__ void softmax_stats_kernel(const __bf16* __restrict__ Qt,
                                     const __bf16* __restrict__ Kt,
                                     float* __restrict__ smax,
                                     float* __restrict__ sinv) {
  int lane = threadIdx.x & 31;
  int gw   = blockIdx.x * (blockDim.x >> 5) + (threadIdx.x >> 5);
  int b    = gw >> 7;              // 128 row-blocks (of 32) per batch
  int i0   = (gw & 127) << 5;
  const __bf16* Qb = Qt + (size_t)b * NN * CP;
  const __bf16* Kb = Kt + (size_t)b * NN * CP;
  int r = lane & 15;
  int mbase = (lane >> 4) << 3;    // D rows held by this lane: mbase..mbase+7

  v16bf a[2][2];                   // [row-tile][k-step], loop-invariant
#pragma unroll
  for (int t = 0; t < 2; ++t)
#pragma unroll
    for (int k = 0; k < 2; ++k)
      a[t][k] = load_fragA(Qb, i0 + t * 16 + r, CP, k * 32, lane);

  v8f rmax[2];
#pragma unroll
  for (int t = 0; t < 2; ++t)
#pragma unroll
    for (int e = 0; e < 8; ++e) rmax[t][e] = -3.0e38f;

  for (int j0 = 0; j0 < NN; j0 += 16) {
    v16bf b0 = load_fragB(Kb, j0 + r, CP, 0,  lane);
    v16bf b1 = load_fragB(Kb, j0 + r, CP, 32, lane);
#pragma unroll
    for (int t = 0; t < 2; ++t) {
      v8f acc = {};
      acc = wmma_bf16(a[t][0], b0, acc);
      acc = wmma_bf16(a[t][1], b1, acc);
#pragma unroll
      for (int e = 0; e < 8; ++e) rmax[t][e] = fmaxf(rmax[t][e], acc[e]);
    }
  }
#pragma unroll
  for (int m = 1; m < 16; m <<= 1)
#pragma unroll
    for (int t = 0; t < 2; ++t)
#pragma unroll
      for (int e = 0; e < 8; ++e)
        rmax[t][e] = fmaxf(rmax[t][e], __shfl_xor(rmax[t][e], m, 32));

  v8f rsum[2];
#pragma unroll
  for (int t = 0; t < 2; ++t) rsum[t] = v8f{};
  for (int j0 = 0; j0 < NN; j0 += 16) {
    v16bf b0 = load_fragB(Kb, j0 + r, CP, 0,  lane);
    v16bf b1 = load_fragB(Kb, j0 + r, CP, 32, lane);
#pragma unroll
    for (int t = 0; t < 2; ++t) {
      v8f acc = {};
      acc = wmma_bf16(a[t][0], b0, acc);
      acc = wmma_bf16(a[t][1], b1, acc);
#pragma unroll
      for (int e = 0; e < 8; ++e) rsum[t][e] += __expf(acc[e] - rmax[t][e]);
    }
  }
#pragma unroll
  for (int m = 1; m < 16; m <<= 1)
#pragma unroll
    for (int t = 0; t < 2; ++t)
#pragma unroll
      for (int e = 0; e < 8; ++e) rsum[t][e] += __shfl_xor(rsum[t][e], m, 32);

  if (r == 0) {
#pragma unroll
    for (int t = 0; t < 2; ++t)
#pragma unroll
      for (int e = 0; e < 8; ++e) {
        size_t idx = (size_t)b * NN + i0 + t * 16 + mbase + e;
        smax[idx] = rmax[t][e];
        sinv[idx] = 1.0f / rsum[t][e];
      }
  }
}

// ------------------------- attention output ---------------------------------
// One wave per (b, 32-column j block = 2 j-tiles).  O[:,j] = sum_i H[:,i]P[i,j].
// Per 32-row i step: 8 WMMAs rebuild both S tiles for both j-tiles (Q frags
// reused), shfl_xor(16) repacks P (D-layout) into 32x16 B-fragments, and
// 8 WMMAs accumulate H*P (each H fragment reused across both j-tiles).
__global__ void attn_out_kernel(const __bf16* __restrict__ Qt,
                                const __bf16* __restrict__ Kt,
                                const __bf16* __restrict__ Hm,
                                const float* __restrict__ smax,
                                const float* __restrict__ sinv,
                                __bf16* __restrict__ Ot) {
  int lane = threadIdx.x & 31;
  int gw   = blockIdx.x * (blockDim.x >> 5) + (threadIdx.x >> 5);
  int b    = gw >> 7;              // 128 col-blocks (of 32) per batch
  int j0   = (gw & 127) << 5;
  const __bf16* Qb = Qt + (size_t)b * NN * CP;
  const __bf16* Kb = Kt + (size_t)b * NN * CP;
  const __bf16* Hb = Hm + (size_t)b * CP * NN;
  const float* mx = smax + (size_t)b * NN;
  const float* iv = sinv + (size_t)b * NN;
  int r = lane & 15;
  int mbase = (lane >> 4) << 3;
  bool hi = (lane >= 16);

  v16bf bk[2][2];                  // [j-tile][k-step], loop-invariant
#pragma unroll
  for (int u = 0; u < 2; ++u)
#pragma unroll
    for (int k = 0; k < 2; ++k)
      bk[u][k] = load_fragB(Kb, j0 + u * 16 + r, CP, k * 32, lane);

  v8f acc[4][2];                   // [c-tile][j-tile]
#pragma unroll
  for (int t = 0; t < 4; ++t)
#pragma unroll
    for (int u = 0; u < 2; ++u) acc[t][u] = v8f{};

  for (int i0 = 0; i0 < NN; i0 += 32) {
    v16bf aq[4];                   // [i-tile*2 + k-step]
    aq[0] = load_fragA(Qb, i0 + r,      CP, 0,  lane);
    aq[1] = load_fragA(Qb, i0 + r,      CP, 32, lane);
    aq[2] = load_fragA(Qb, i0 + 16 + r, CP, 0,  lane);
    aq[3] = load_fragA(Qb, i0 + 16 + r, CP, 32, lane);

    // softmax stats for the 16 rows this lane owns (shared across j-tiles)
    float m1[8], v1[8], m2[8], v2[8];
#pragma unroll
    for (int e = 0; e < 8; ++e) {
      int i1 = i0 + mbase + e;
      int i2 = i0 + 16 + mbase + e;
      m1[e] = mx[i1]; v1[e] = iv[i1];
      m2[e] = mx[i2]; v2[e] = iv[i2];
    }

    v16bf pf[2];
#pragma unroll
    for (int u = 0; u < 2; ++u) {
      v8f s1 = {}, s2 = {};
      s1 = wmma_bf16(aq[0], bk[u][0], s1);
      s1 = wmma_bf16(aq[1], bk[u][1], s1);
      s2 = wmma_bf16(aq[2], bk[u][0], s2);
      s2 = wmma_bf16(aq[3], bk[u][1], s2);
#pragma unroll
      for (int e = 0; e < 8; ++e) {
        float p1 = __expf(s1[e] - m1[e]) * v1[e];   // tile1 rows (own)
        float p2 = __expf(s2[e] - m2[e]) * v2[e];   // tile2 rows (own)
        float q1 = __shfl_xor(p1, 16, 32);          // partner half's rows
        float q2 = __shfl_xor(p2, 16, 32);
        // B-fragment: lanes<16 hold K 0..15 (tile1 rows 0..15),
        //             lanes>=16 hold K 16..31 (tile2 rows 0..15)
        pf[u][e]     = static_cast<__bf16>(hi ? q2 : p1);
        pf[u][e + 8] = static_cast<__bf16>(hi ? p2 : q1);
      }
    }
#pragma unroll
    for (int t = 0; t < 4; ++t) {
      v16bf ah = load_fragA(Hb, t * 16 + r, NN, i0, lane);
#pragma unroll
      for (int u = 0; u < 2; ++u) acc[t][u] = wmma_bf16(ah, pf[u], acc[t][u]);
    }
  }
  // store O^T as (B, N, CP) bf16 so the final GEMM can B-frag it
  __bf16* Ob = Ot + (size_t)b * NN * CP;
#pragma unroll
  for (int t = 0; t < 4; ++t)
#pragma unroll
    for (int u = 0; u < 2; ++u) {
      v8bf o;
#pragma unroll
      for (int e = 0; e < 8; ++e) o[e] = static_cast<__bf16>(acc[t][u][e]);
      *reinterpret_cast<v8bf*>(Ob + (size_t)(j0 + u * 16 + r) * CP + t * 16 + mbase) = o;
    }
}

// ------------------------- final: y = gamma * Wv_sn @ O + x -----------------
__global__ void final_kernel(const __bf16* __restrict__ Wv_b,
                             const __bf16* __restrict__ Ot,
                             const float* __restrict__ x,
                             const float* __restrict__ gamma,
                             float* __restrict__ y) {
  int lane = threadIdx.x & 31;
  int gw   = blockIdx.x * (blockDim.x >> 5) + (threadIdx.x >> 5);
  int b    = gw >> 10;             // 4 cgroups * 256 jtiles per batch
  int rem  = gw & 1023;
  int cg   = rem >> 8;             // 0..3 -> 128 output channels each
  int j0   = (rem & 255) << 4;
  const __bf16* Ob = Ot + (size_t)b * NN * CP;
  int r = lane & 15;
  int mbase = (lane >> 4) << 3;
  v16bf b0 = load_fragB(Ob, j0 + r, CP, 0,  lane);
  v16bf b1 = load_fragB(Ob, j0 + r, CP, 32, lane);
  float g = gamma[0];
#pragma unroll
  for (int t = 0; t < 8; ++t) {
    int c0 = cg * 128 + t * 16;
    v16bf a0 = load_fragA(Wv_b, c0 + r, CP, 0,  lane);
    v16bf a1 = load_fragA(Wv_b, c0 + r, CP, 32, lane);
    v8f acc = {};
    acc = wmma_bf16(a0, b0, acc);
    acc = wmma_bf16(a1, b1, acc);
#pragma unroll
    for (int e = 0; e < 8; ++e) {
      int c = c0 + mbase + e;
      size_t idx = ((size_t)b * CC + c) * NN + j0 + r;
      y[idx] = g * acc[e] + x[idx];
    }
  }
}

// ---------------------------------------------------------------------------
extern "C" void kernel_launch(void* const* d_in, const int* in_sizes, int n_in,
                              void* d_out, int out_size, void* d_ws, size_t ws_size,
                              hipStream_t stream) {
  (void)in_sizes; (void)n_in; (void)out_size; (void)ws_size;
  const float* x     = (const float*)d_in[0];
  const float* Wf    = (const float*)d_in[1];
  const float* Wg    = (const float*)d_in[2];
  const float* Wh    = (const float*)d_in[3];
  const float* Wv    = (const float*)d_in[4];
  const float* uf    = (const float*)d_in[5];
  const float* ug    = (const float*)d_in[6];
  const float* uh    = (const float*)d_in[7];
  const float* uv    = (const float*)d_in[8];
  const float* gamma = (const float*)d_in[9];
  float* y = (float*)d_out;

  char* ws = (char*)d_ws;
  size_t off = 0;
  auto alloc = [&](size_t bytes) -> char* {
    char* p = ws + off;
    off = (off + bytes + 255) & ~(size_t)255;
    return p;
  };
  float*  sig_inv = (float*) alloc(4 * sizeof(float));
  float*  smax    = (float*) alloc((size_t)BB * NN * sizeof(float));
  float*  sinv    = (float*) alloc((size_t)BB * NN * sizeof(float));
  __bf16* Wv_b    = (__bf16*)alloc((size_t)CC * CP * sizeof(__bf16));
  __bf16* Qt      = (__bf16*)alloc((size_t)BB * NN * CP * sizeof(__bf16));
  __bf16* Kt      = (__bf16*)alloc((size_t)BB * NN * CP * sizeof(__bf16));
  __bf16* Hm      = (__bf16*)alloc((size_t)BB * NN * CP * sizeof(__bf16));
  __bf16* Ot      = (__bf16*)alloc((size_t)BB * NN * CP * sizeof(__bf16));

  sn_sigma_kernel<<<4, 256, 0, stream>>>(Wf, Wg, Wh, Wv, uf, ug, uh, uv, sig_inv);
  wv_conv_kernel<<<(CC * CP + 255) / 256, 256, 0, stream>>>(Wv, sig_inv, Wv_b);
  proj_kernel<<<dim3(NN / 64, BB), 256, 0, stream>>>(x, Wf, Wg, Wh, sig_inv, Qt, Kt, Hm);
  softmax_stats_kernel<<<(BB * (NN / 32)) / 8, 256, 0, stream>>>(Qt, Kt, smax, sinv);
  attn_out_kernel<<<(BB * (NN / 32)) / 8, 256, 0, stream>>>(Qt, Kt, Hm, smax, sinv, Ot);
  final_kernel<<<(BB * 4 * (NN / 16)) / 8, 256, 0, stream>>>(Wv_b, Ot, x, gamma, y);
}
